// SwinTransformerBlock_13735305413348
// MI455X (gfx1250) — matrix operand order repaired and checked by reading, not compiled, not run on hardware
//
#include <hip/hip_runtime.h>
#include <hip/hip_bf16.h>
#include <math.h>

typedef _Float16 v8h  __attribute__((ext_vector_type(8)));
typedef _Float16 v16h __attribute__((ext_vector_type(16)));
typedef float    v8f  __attribute__((ext_vector_type(8)));

#define BDIM   16
#define HDIM   64
#define WDIM   64
#define CDIM   512
#define NHEAD  16
#define WSZ    8
#define SSZ    4
#define NTOK   64          // tokens per window
#define HD     32          // head dim
#define HIDDEN 2048
#define NWIN   64          // windows per image
#define BW     1024        // BDIM*NWIN
#define MROWS  65536       // BW*NTOK == BDIM*HDIM*WDIM
#define SCALE  0.17677669529663687f   // 32^-0.5
#define PI_F   3.14159265358979f

// ---------------------------------------------------------------------------
// WMMA wrapper (guard host pass)
// ---------------------------------------------------------------------------
__device__ inline v8f wmma_f16(v16h a, v16h b, v8f c) {
#if defined(__gfx1250__)
  return __builtin_amdgcn_wmma_f32_16x16x32_f16(false, a, false, b, (short)0, c,
                                                false, false);
#else
  return c;
#endif
}

// ---------------------------------------------------------------------------
// TDM: 2D tile load Global -> LDS through the Tensor Data Mover.
// Descriptor layout per CDNA5 ISA (D# group0/group1).
// data_size = 2 bytes (f16). tileX elements contiguous, tileY rows with
// row stride strideElems (in elements). Wave-level op (EXEC ignored).
// ---------------------------------------------------------------------------
__device__ inline void tdm_load_2d_f16(unsigned ldsAddr, const void* gsrc,
                                       unsigned tensorD0, unsigned tensorD1,
                                       unsigned tileX, unsigned tileY,
                                       unsigned strideElems) {
#if defined(__gfx1250__)
  typedef unsigned u32x4 __attribute__((ext_vector_type(4)));
  typedef int      i32x8 __attribute__((ext_vector_type(8)));
  typedef int      i32x4 __attribute__((ext_vector_type(4)));
  unsigned long long ga = (unsigned long long)gsrc;
  u32x4 g0;
  g0.x = 1u;                                            // count=1, user D#
  g0.y = ldsAddr;                                       // lds_addr
  g0.z = (unsigned)(ga & 0xFFFFFFFFu);                  // global_addr[31:0]
  g0.w = (unsigned)((ga >> 32) & 0x01FFFFFFu) | (2u << 30);  // addr[56:32] | type=2
  i32x8 g1;
  g1[0] = (int)(1u << 16);                              // data_size=1 (2B), wg_mask=0
  g1[1] = (int)((tensorD0 & 0xFFFFu) << 16);            // tensor_dim0[15:0]
  g1[2] = (int)((tensorD0 >> 16) | ((tensorD1 & 0xFFFFu) << 16)); // d0 hi | d1 lo
  g1[3] = (int)((tensorD1 >> 16) | (tileX << 16));      // d1 hi | tile_dim0
  g1[4] = (int)(tileY & 0xFFFFu);                       // tile_dim1 (tile_dim2=0)
  g1[5] = (int)strideElems;                             // tensor_dim0_stride[31:0]
  g1[6] = 0;                                            // stride hi / dim1_stride lo
  g1[7] = 0;
  i32x4 gz = {0, 0, 0, 0};
#if __clang_major__ >= 23
  i32x8 gz8 = {0, 0, 0, 0, 0, 0, 0, 0};
  __builtin_amdgcn_tensor_load_to_lds(g0, g1, gz, gz, gz8, 0);
#else
  __builtin_amdgcn_tensor_load_to_lds(g0, g1, gz, gz, 0);
#endif
#else
  (void)ldsAddr; (void)gsrc; (void)tensorD0; (void)tensorD1;
  (void)tileX; (void)tileY; (void)strideElems;
#endif
}

__device__ inline void tdm_wait0() {
#if defined(__gfx1250__)
  __builtin_amdgcn_s_wait_tensorcnt(0);
#endif
}

__device__ inline unsigned lds_off(const void* p) {
  return (unsigned)(uintptr_t)p;   // generic LDS addr: low 32 bits = LDS offset
}

// A fragment (16x32 f16): lane&15 = M row; chunk0 halves t=0..7 -> K = 8*(lane>=16)+t,
// chunk1 halves t=8..15 -> K = 8*(lane>=16)+16+t-8.
__device__ inline v16h load_a_frag(const _Float16* s, int stride, int rowBase,
                                   int kBase, int lane) {
  const _Float16* p = s + (size_t)(rowBase + (lane & 15)) * stride + kBase + 8 * (lane >> 4);
  v8h lo = *(const v8h*)p;
  v8h hi = *(const v8h*)(p + 16);
  return __builtin_shufflevector(lo, hi, 0,1,2,3,4,5,6,7,8,9,10,11,12,13,14,15);
}

// B fragment (32x16 f16) from an LDS tile stored transposed [N][K]:
// lane&15 = N col; halves t=0..7 -> K = 8*(lane>=16)+t, t=8..15 -> K = 8*(lane>=16)+16+t-8.
__device__ inline v16h load_b_frag(const _Float16* sT, int stride, int nBase,
                                   int kBase, int lane) {
  const _Float16* p = sT + (size_t)(nBase + (lane & 15)) * stride + kBase + 8 * (lane >> 4);
  v8h lo = *(const v8h*)p;
  v8h hi = *(const v8h*)(p + 16);
  return __builtin_shufflevector(lo, hi, 0,1,2,3,4,5,6,7,8,9,10,11,12,13,14,15);
}

__device__ inline float waveSum(float v) {
  for (int o = 16; o > 0; o >>= 1) v += __shfl_xor(v, o, 32);
  return v;
}

// ---------------------------------------------------------------------------
// Kernel 1: convert all weights f32 -> f16
// ---------------------------------------------------------------------------
__global__ __launch_bounds__(256) void convert_weights(
    const float* __restrict__ wqkv, const float* __restrict__ wproj,
    const float* __restrict__ w1,   const float* __restrict__ w2,
    _Float16* __restrict__ oqkv, _Float16* __restrict__ oproj,
    _Float16* __restrict__ o1,   _Float16* __restrict__ o2) {
  int g = blockIdx.x * 256 + threadIdx.x;
  if (g < 786432)             oqkv[g]            = (_Float16)wqkv[g];
  else if (g < 1048576)       oproj[g - 786432]  = (_Float16)wproj[g - 786432];
  else if (g < 2097152)       o1[g - 1048576]    = (_Float16)w1[g - 1048576];
  else if (g < 3145728)       o2[g - 2097152]    = (_Float16)w2[g - 2097152];
}

// ---------------------------------------------------------------------------
// Kernel 2: A_phi[h][n][m]  (16 x 64 x 64)
// ---------------------------------------------------------------------------
__global__ __launch_bounds__(256) void aphi_kernel(
    const float* __restrict__ a_p, const float* __restrict__ b_p,
    float* __restrict__ aphi) {
  int g = blockIdx.x * 256 + threadIdx.x;          // 65536 total
  int h = g >> 12, n = (g >> 6) & 63, m = g & 63;
  float az = (float)((n & 7) - (m & 7)) * (2.0f * PI_F / 64.0f);
  float acc = 0.0f;
  for (int p = 0; p < 4; ++p) {
    float ang = az * (float)(p + 1);
    acc += cosf(ang) * a_p[(p + 1) * NHEAD + h] + sinf(ang) * b_p[p * NHEAD + h];
  }
  aphi[g] = acc * 0.25f + a_p[h];
}

// ---------------------------------------------------------------------------
// Kernel 3: LN1 + roll(-4,-4) + window partition -> hw f16 (BW,64,512), dw f32
// ---------------------------------------------------------------------------
__global__ __launch_bounds__(256) void prep_ln1(
    const float* __restrict__ x, const float* __restrict__ D,
    const float* __restrict__ g1, const float* __restrict__ be1,
    _Float16* __restrict__ hw, float* __restrict__ dwv) {
  int wave = threadIdx.x >> 5, lane = threadIdx.x & 31;
  int T = blockIdx.x * 8 + wave;                   // 0..65535
  int bw = T >> 6, n = T & 63;
  int batch = bw >> 6, win = bw & 63;
  int hh = (((win >> 3) << 3) + (n >> 3) + SSZ) & 63;
  int ww = (((win & 7) << 3) + (n & 7) + SSZ) & 63;
  size_t tok = (size_t)batch * 4096 + hh * 64 + ww;
  size_t src = tok * CDIM;
  float vals[16];
  float s = 0.f;
#pragma unroll
  for (int e = 0; e < 16; ++e) { vals[e] = x[src + lane + 32 * e]; s += vals[e]; }
  float mean = waveSum(s) * (1.0f / CDIM);
  float vs = 0.f;
#pragma unroll
  for (int e = 0; e < 16; ++e) { float d = vals[e] - mean; vs += d * d; }
  float rstd = rsqrtf(waveSum(vs) * (1.0f / CDIM) + 1e-5f);
#pragma unroll
  for (int e = 0; e < 16; ++e) {
    int c = lane + 32 * e;
    hw[(size_t)T * CDIM + c] = (_Float16)((vals[e] - mean) * rstd * g1[c] + be1[c]);
  }
  if (lane == 0) dwv[T] = D[tok];
}

// ---------------------------------------------------------------------------
// Kernel 7: LN2 over x2 -> f16
// ---------------------------------------------------------------------------
__global__ __launch_bounds__(256) void ln2_kernel(
    const float* __restrict__ x2, const float* __restrict__ g2,
    const float* __restrict__ be2, _Float16* __restrict__ h2in) {
  int wave = threadIdx.x >> 5, lane = threadIdx.x & 31;
  size_t T = (size_t)blockIdx.x * 8 + wave;
  size_t src = T * CDIM;
  float vals[16];
  float s = 0.f;
#pragma unroll
  for (int e = 0; e < 16; ++e) { vals[e] = x2[src + lane + 32 * e]; s += vals[e]; }
  float mean = waveSum(s) * (1.0f / CDIM);
  float vs = 0.f;
#pragma unroll
  for (int e = 0; e < 16; ++e) { float d = vals[e] - mean; vs += d * d; }
  float rstd = rsqrtf(waveSum(vs) * (1.0f / CDIM) + 1e-5f);
#pragma unroll
  for (int e = 0; e < 16; ++e) {
    int c = lane + 32 * e;
    h2in[src + c] = (_Float16)((vals[e] - mean) * rstd * g2[c] + be2[c]);
  }
}

// ---------------------------------------------------------------------------
// Tiled WMMA GEMM: C[M,N] = A[M,K](f16) * W[K,N](f16) + bias, fused epilogue.
// A tile is DMA'd by the Tensor Data Mover; B tile staged transposed by VALU.
// MODE 0: QKV split  MODE 1: proj+scatter+residual  MODE 2: MLP1+GELU
// MODE 3: MLP2+residual -> out
// ---------------------------------------------------------------------------
template <int MODE>
__global__ __launch_bounds__(128) void gemm64(
    const _Float16* __restrict__ A, const _Float16* __restrict__ Wt,
    const float* __restrict__ bias, int K, int N,
    _Float16* __restrict__ qb, _Float16* __restrict__ kb, _Float16* __restrict__ vb,
    const float* __restrict__ xin, float* __restrict__ xout,
    _Float16* __restrict__ hout) {
  __shared__ __align__(32) _Float16 sA[64 * 32];   // [m][k] (TDM destination)
  __shared__ __align__(32) _Float16 sB[64 * 32];   // transposed [n][k]
  const int tid = threadIdx.x;
  const int wave = tid >> 5, lane = tid & 31;
  const int m0 = blockIdx.x * 64;
  const int n0 = blockIdx.y * 64;
  const unsigned ldsA = lds_off(&sA[0]);

  v8f acc[4];
#pragma unroll
  for (int j = 0; j < 4; ++j) acc[j] = (v8f){};

  for (int k0 = 0; k0 < K; k0 += 32) {
    // --- A tile via TDM (wave 0 issues; 64 rows x 32 halves, row stride K)
    if (wave == 0) {
      tdm_load_2d_f16(ldsA, &A[(size_t)m0 * K + k0],
                      (unsigned)K, 65536u, 32u, 64u, (unsigned)K);
    }
    // --- stage B tile transposed: [n][k]
#pragma unroll
    for (int p = 0; p < 2; ++p) {
      int krow = (tid >> 3) + 16 * p;
      int nc8  = (tid & 7) << 3;
      const _Float16* src = &Wt[(size_t)(k0 + krow) * N + n0 + nc8];
      v8h tmp = *(const v8h*)src;
#pragma unroll
      for (int e = 0; e < 8; ++e) sB[(nc8 + e) * 32 + krow] = tmp[e];
      if (k0 + 32 < K) __builtin_prefetch(src + (size_t)32 * N, 0, 3);
    }
    if (wave == 0) tdm_wait0();    // TENSORcnt==0 before the barrier
    __syncthreads();

    v16h af = load_a_frag(sA, 32, 16 * wave, 0, lane);
#pragma unroll
    for (int j = 0; j < 4; ++j) {
      v16h bf = load_b_frag(sB, 32, 16 * j, 0, lane);
      acc[j] = wmma_f16(af, bf, acc[j]);
    }
    __syncthreads();
  }

  // ---- epilogue (C/D layout: reg i, lane l -> M = i + 8*(l>=16), N = l&15)
  const int mBase = m0 + 16 * wave;
#pragma unroll
  for (int j = 0; j < 4; ++j) {
#pragma unroll
    for (int i = 0; i < 8; ++i) {
      int m   = mBase + i + 8 * (lane >> 4);
      int col = n0 + 16 * j + (lane & 15);
      float val = acc[j][i] + bias[col];
      if (MODE == 0) {
        int s = col >> 9, rem = col & 511;
        int h = rem >> 5, d = rem & 31;
        int b = m >> 6, n = m & 63;
        _Float16* dst = (s == 0) ? qb : (s == 1) ? kb : vb;
        dst[((((size_t)b * NHEAD + h) * NTOK) + n) * HD + d] = (_Float16)val;
      } else if (MODE == 1) {
        int b = m >> 6, n = m & 63;
        int batch = b >> 6, win = b & 63;
        int hh = (((win >> 3) << 3) + (n >> 3) + SSZ) & 63;
        int ww = (((win & 7) << 3) + (n & 7) + SSZ) & 63;
        size_t tok = (size_t)batch * 4096 + hh * 64 + ww;
        xout[tok * CDIM + col] = xin[tok * CDIM + col] + val;
      } else if (MODE == 2) {
        float g = 0.5f * val * (1.0f + erff(val * 0.70710678118f));
        hout[(size_t)m * HIDDEN + col] = (_Float16)g;
      } else {  // MODE 3
        xout[(size_t)m * CDIM + col] = val + xin[(size_t)m * CDIM + col];
      }
    }
  }
}

// ---------------------------------------------------------------------------
// Kernel 5: windowed attention, one block (128 thr = 4 waves) per (window,head)
// q/k tiles DMA'd by TDM; v transposed by VALU staging.
// ---------------------------------------------------------------------------
__global__ __launch_bounds__(128) void attn_kernel(
    const _Float16* __restrict__ q, const _Float16* __restrict__ k,
    const _Float16* __restrict__ v, const float* __restrict__ dwv,
    const float* __restrict__ aphi, const float* __restrict__ a_r,
    const float* __restrict__ b_r, _Float16* __restrict__ out) {
  __shared__ __align__(32) _Float16 sq[64 * 32];   // [n][d]
  __shared__ __align__(32) _Float16 sk[64 * 32];   // [m][d] (B transposed layout)
  __shared__ __align__(32) _Float16 svT[32 * 64];  // [d][m]
  __shared__ float sdw[64];
  __shared__ float sS[64 * 64];                    // scores f32
  __shared__ __align__(32) _Float16 sp[64 * 64];   // probs f16 [n][m]

  const int tid = threadIdx.x;
  const int wave = tid >> 5, lane = tid & 31;
  const int bh = blockIdx.x;
  const int b = bh >> 4, h = bh & 15;
  const size_t base = ((size_t)b * NHEAD + h) * NTOK * HD;

  // q/k: contiguous 2048-half tiles via TDM (wave 0)
  if (wave == 0) {
    tdm_load_2d_f16(lds_off(&sq[0]), &q[base], 2048u, 1u, 2048u, 1u, 2048u);
    tdm_load_2d_f16(lds_off(&sk[0]), &k[base], 2048u, 1u, 2048u, 1u, 2048u);
  }
  // v transposed + dw row (all waves)
  {
    int mrow = tid >> 1, dbase = (tid & 1) << 4;
    const _Float16* vsrc = &v[base + (size_t)mrow * HD + dbase];
#pragma unroll
    for (int e = 0; e < 16; ++e) svT[(dbase + e) * 64 + mrow] = vsrc[e];
    if (tid < 64) sdw[tid] = dwv[b * 64 + tid];
  }
  // per-thread radial coefficients
  float ar0 = a_r[h];
  float arp[4], brp[4];
#pragma unroll
  for (int p = 0; p < 4; ++p) {
    arp[p] = a_r[(p + 1) * NHEAD + h];
    brp[p] = b_r[p * NHEAD + h];
  }
  if (wave == 0) tdm_wait0();
  __syncthreads();

  // ---- scores: wave strip rows nb..nb+15, all 64 cols
  const int nb = 16 * wave;
  v8f accS[4];
  {
    v16h af = load_a_frag(sq, 32, nb, 0, lane);
#pragma unroll
    for (int j = 0; j < 4; ++j) {
      accS[j] = (v8f){};
      v16h bf = load_b_frag(sk, 32, 16 * j, 0, lane);
      accS[j] = wmma_f16(af, bf, accS[j]);
    }
  }
  // bias + mask epilogue, write f32 scores to LDS
  const int win = b & 63, wh = win >> 3, wwc = win & 7;
#pragma unroll
  for (int j = 0; j < 4; ++j) {
#pragma unroll
    for (int i = 0; i < 8; ++i) {
      int n = nb + i + 8 * (lane >> 4);
      int m = 16 * j + (lane & 15);
      float val = accS[j][i] * SCALE + aphi[(size_t)h * 4096 + n * 64 + m];
      float radD = sdw[m] - sdw[n];
      float racc = 0.0f;
#pragma unroll
      for (int p = 0; p < 4; ++p) {
        float ang = radD * (PI_F * (float)(p + 1));
        racc += cosf(ang) * arp[p] + sinf(ang) * brp[p];
      }
      val += racc * 0.25f + ar0;
      // shift mask
      int hgN = wh * 8 + (n >> 3), wgN = wwc * 8 + (n & 7);
      int hgM = wh * 8 + (m >> 3), wgM = wwc * 8 + (m & 7);
      int ln_ = (hgN < 56 ? 0 : hgN < 60 ? 1 : 2) * 3 + (wgN < 56 ? 0 : wgN < 60 ? 1 : 2);
      int lm_ = (hgM < 56 ? 0 : hgM < 60 ? 1 : 2) * 3 + (wgM < 56 ? 0 : wgM < 60 ? 1 : 2);
      if (ln_ != lm_) val -= 100.0f;
      sS[n * 64 + m] = val;
    }
  }
  __syncthreads();

  // ---- softmax per row (threads 0..63)
  if (tid < 64) {
    float* row = &sS[tid * 64];
    float mx = -1e30f;
#pragma unroll 4
    for (int m = 0; m < 64; ++m) mx = fmaxf(mx, row[m]);
    float sum = 0.0f;
    float e[64];
#pragma unroll 4
    for (int m = 0; m < 64; ++m) { e[m] = expf(row[m] - mx); sum += e[m]; }
    float inv = 1.0f / sum;
#pragma unroll 4
    for (int m = 0; m < 64; ++m) sp[tid * 64 + m] = (_Float16)(e[m] * inv);
  }
  __syncthreads();

  // ---- out = P(64x64) @ V(64x32): per wave 16 rows x 32 cols
  v8f accO[2];
  accO[0] = (v8f){}; accO[1] = (v8f){};
#pragma unroll
  for (int kc = 0; kc < 2; ++kc) {
    v16h af = load_a_frag(sp, 64, nb, 32 * kc, lane);
#pragma unroll
    for (int jc = 0; jc < 2; ++jc) {
      v16h bf = load_b_frag(svT, 64, 16 * jc, 32 * kc, lane);
      accO[jc] = wmma_f16(af, bf, accO[jc]);
    }
  }
#pragma unroll
  for (int jc = 0; jc < 2; ++jc) {
#pragma unroll
    for (int i = 0; i < 8; ++i) {
      int n = nb + i + 8 * (lane >> 4);
      int d = 16 * jc + (lane & 15);
      out[((size_t)b * NTOK + n) * CDIM + h * HD + d] = (_Float16)accO[jc][i];
    }
  }
}

// ---------------------------------------------------------------------------
// launch
// ---------------------------------------------------------------------------
extern "C" void kernel_launch(void* const* d_in, const int* in_sizes, int n_in,
                              void* d_out, int out_size, void* d_ws, size_t ws_size,
                              hipStream_t stream) {
  const float* x     = (const float*)d_in[0];
  const float* D     = (const float*)d_in[1];
  const float* g1    = (const float*)d_in[2];
  const float* be1   = (const float*)d_in[3];
  const float* Wqkv  = (const float*)d_in[4];
  const float* bqkv  = (const float*)d_in[5];
  const float* Wproj = (const float*)d_in[6];
  const float* bproj = (const float*)d_in[7];
  const float* a_p   = (const float*)d_in[8];
  const float* b_p   = (const float*)d_in[9];
  const float* a_r   = (const float*)d_in[10];
  const float* b_r   = (const float*)d_in[11];
  const float* g2    = (const float*)d_in[12];
  const float* be2   = (const float*)d_in[13];
  const float* W1    = (const float*)d_in[14];
  const float* b1    = (const float*)d_in[15];
  const float* W2    = (const float*)d_in[16];
  const float* b2    = (const float*)d_in[17];
  float* out = (float*)d_out;

  char* ws = (char*)d_ws;
  // region A (268 MB): hw/attn_out + q/k/v early, mid later
  _Float16* hw   = (_Float16*)(ws);                        // 67,108,864 B
  _Float16* qb   = (_Float16*)(ws + 67108864ull);
  _Float16* kb   = (_Float16*)(ws + 134217728ull);
  _Float16* vb   = (_Float16*)(ws + 201326592ull);
  _Float16* attn_out = hw;                                 // reuse (hw dead)
  _Float16* mid  = (_Float16*)(ws);                        // reuse region A
  float*    x2   = (float*)(ws + 268435456ull);            // 134,217,728 B
  _Float16* h2in = (_Float16*)(ws + 402653184ull);         // 67,108,864 B
  float*    dwv  = (float*)(ws + 469762048ull);            // 262,144 B
  float*    aphi = (float*)(ws + 470024192ull);            // 262,144 B
  _Float16* wqkv16  = (_Float16*)(ws + 470286336ull);      // 1,572,864 B
  _Float16* wproj16 = (_Float16*)(ws + 471859200ull);      // 524,288 B
  _Float16* w1_16   = (_Float16*)(ws + 472383488ull);      // 2,097,152 B
  _Float16* w2_16   = (_Float16*)(ws + 474480640ull);      // 2,097,152 B

  convert_weights<<<12288, 256, 0, stream>>>(Wqkv, Wproj, W1, W2,
                                             wqkv16, wproj16, w1_16, w2_16);
  aphi_kernel<<<256, 256, 0, stream>>>(a_p, b_p, aphi);
  prep_ln1<<<8192, 256, 0, stream>>>(x, D, g1, be1, hw, dwv);

  // QKV: (65536,512) x (512,1536)
  gemm64<0><<<dim3(1024, 24), 128, 0, stream>>>(hw, wqkv16, bqkv, 512, 1536,
                                                qb, kb, vb, nullptr, nullptr, nullptr);
  // attention
  attn_kernel<<<16384, 128, 0, stream>>>(qb, kb, vb, dwv, aphi, a_r, b_r, attn_out);
  // proj + reverse window/roll + residual -> x2
  gemm64<1><<<dim3(1024, 8), 128, 0, stream>>>(attn_out, wproj16, bproj, 512, 512,
                                               nullptr, nullptr, nullptr, x, x2, nullptr);
  ln2_kernel<<<8192, 256, 0, stream>>>(x2, g2, be2, h2in);
  // MLP1 + GELU
  gemm64<2><<<dim3(1024, 32), 128, 0, stream>>>(h2in, w1_16, b1, 512, 2048,
                                                nullptr, nullptr, nullptr, nullptr, nullptr, mid);
  // MLP2 + residual -> out
  gemm64<3><<<dim3(1024, 8), 128, 0, stream>>>(mid, w2_16, b2, 2048, 512,
                                               nullptr, nullptr, nullptr, x2, out, nullptr);
  (void)in_sizes; (void)n_in; (void)out_size; (void)ws_size;
}
